// GraphEmbedding_54400055771687
// MI455X (gfx1250) — compile-verified, matrix operand order checked by reading, not run on hardware
//
#include <hip/hip_runtime.h>
#include <math.h>

// ---------------------------------------------------------------------------
// CDNA5 (gfx1250) WMMA types & helpers
// ---------------------------------------------------------------------------
typedef __attribute__((ext_vector_type(16))) __bf16 v16bf;
typedef __attribute__((ext_vector_type(8)))  float  v8f;

static_assert(sizeof(v16bf) == 32, "v16bf must be 32 bytes");

union BF16x2 { unsigned int u; __bf16 h[2]; };
union BFRAG  { uint4 q[2]; v16bf v; };

__device__ __forceinline__ v8f wmma_bf16(v16bf a, v16bf b, v8f c) {
  // D = A(16x32 bf16) x B(32x16 bf16) + C(16x16 f32)
  return __builtin_amdgcn_wmma_f32_16x16x32_bf16(
      /*neg_a=*/false, a, /*neg_b=*/false, b,
      /*c_mod=*/(short)0, c, /*reuse_a=*/false, /*reuse_b=*/false);
}

// B fragments are pre-swizzled: fragment = 512 contiguous bf16, lane l owns
// elements [l*16, l*16+16) -> two 16B vector loads per lane.
__device__ __forceinline__ v16bf load_bfrag(const __bf16* base, int lane) {
  BFRAG f;
  const uint4* p = reinterpret_cast<const uint4*>(base) + lane * 2;
  f.q[0] = p[0];
  f.q[1] = p[1];
  return f.v;
}

// A fragment (16x32 bf16) from an LDS row-major tile; `row` points at the
// start of this lane's M-row.  Pair p covers K = kbase + koff(p) .. +1.
__device__ __forceinline__ v16bf load_afrag_lds(const __bf16* row, int kbase, int half) {
  v16bf a;
#pragma unroll
  for (int p = 0; p < 8; ++p) {
    int k = kbase + ((p >> 2) << 4) + ((p & 3) << 1) + (half << 3);
    BF16x2 c;
    c.u = *reinterpret_cast<const unsigned int*>(row + k);
    a[2 * p]     = c.h[0];
    a[2 * p + 1] = c.h[1];
  }
  return a;
}

__device__ __forceinline__ float fast_sigmoid(float x) {
  return 1.0f / (1.0f + __expf(-x));
}

// ---------------------------------------------------------------------------
// Problem constants
// ---------------------------------------------------------------------------
constexpr int Bq   = 256;     // batch of query nodes
constexpr int Kn   = 20;      // neighbors
constexpr int Dd   = 128;     // feature dim
constexpr int FEAT = 484;     // concat feature dim
constexpr int FS   = 520;     // LDS feat row stride (bf16), padded vs 512

// Swizzled-weight region layout (element offsets into bf16 scratch)
constexpr long EL_WQ   = 8L * 16 * 512;    // K=256  N=256
constexpr long EL_WK   = 16L * 16 * 512;   // K=512(pad from 484) N=256
constexpr long EL_WV   = EL_WK;
constexpr long EL_WIH  = 8L * 256 * 512;   // K=256  N=4096
constexpr long EL_WOUT = 36L * 8 * 512;    // K=1152 N=128
constexpr long OFF_WQ   = 0;
constexpr long OFF_WK   = OFF_WQ + EL_WQ;
constexpr long OFF_WV   = OFF_WK + EL_WK;
constexpr long OFF_WIH  = OFF_WV + EL_WV;
constexpr long OFF_WOUT = OFF_WIH + EL_WIH;
constexpr long TOTAL_SW = OFF_WOUT + EL_WOUT;

// ---------------------------------------------------------------------------
// Kernel 0: convert f32 weights -> bf16, pre-swizzled into B-fragment order.
// Fragment (kt, nt): element (l, e) = W[kt*32 + koff(l,e)][nt*16 + (l&15)]
// with koff(l,e) = (e&7) + 8*(l>>4) + 16*(e>>3); zero-padded along K.
// ---------------------------------------------------------------------------
__device__ __forceinline__ void swiz_one(const float* __restrict__ W,
                                         __bf16* __restrict__ dst,
                                         int Ksrc, int N, long idx) {
  int frag = (int)(idx >> 9);
  int r    = (int)(idx & 511);
  int l = r >> 4, e = r & 15;
  int ntiles = N >> 4;
  int kt = frag / ntiles;
  int nt = frag - kt * ntiles;
  int k = (kt << 5) + (e & 7) + ((l >> 4) << 3) + ((e >> 3) << 4);
  int n = (nt << 4) + (l & 15);
  float v = (k < Ksrc) ? W[(long)k * N + n] : 0.0f;
  dst[idx] = (__bf16)v;
}

__global__ void prep_weights(const float* __restrict__ Wq,
                             const float* __restrict__ Wk,
                             const float* __restrict__ Wv,
                             const float* __restrict__ Wih,
                             const float* __restrict__ Wout,
                             __bf16* __restrict__ sw) {
  long idx = (long)blockIdx.x * blockDim.x + threadIdx.x;
  if (idx < EL_WQ)  { swiz_one(Wq,  sw + OFF_WQ,  256,  256,  idx); return; }
  idx -= EL_WQ;
  if (idx < EL_WK)  { swiz_one(Wk,  sw + OFF_WK,  484,  256,  idx); return; }
  idx -= EL_WK;
  if (idx < EL_WV)  { swiz_one(Wv,  sw + OFF_WV,  484,  256,  idx); return; }
  idx -= EL_WV;
  if (idx < EL_WIH) { swiz_one(Wih, sw + OFF_WIH, 256,  4096, idx); return; }
  idx -= EL_WIH;
  if (idx < EL_WOUT){ swiz_one(Wout,sw + OFF_WOUT,1152, 128,  idx); return; }
}

// ---------------------------------------------------------------------------
// Kernel 1: q = [node_feats(ids) , cos(time_b)] @ Wq     (n x 256, f32 out)
// One workgroup per 16-row M tile; 8 waves each own a 32-col N slice.
// ---------------------------------------------------------------------------
__global__ __launch_bounds__(256) void gemm_q(const int* __restrict__ ids,
                                              const float* __restrict__ nr,
                                              const float* __restrict__ ne,
                                              const float* __restrict__ time_b,
                                              const __bf16* __restrict__ Wq_sw,
                                              float* __restrict__ qout) {
  __shared__ int idS[16];
  __shared__ float tcS[128];   // cos(time_b) = time_enc(0), block-wide constant
  int tid = threadIdx.x;
  int wv = tid >> 5, lane = tid & 31, half = lane >> 4, lm = lane & 15;
  int m0 = blockIdx.x * 16;
  if (tid < 16) idS[tid] = ids[m0 + tid];
  if (tid < 128) tcS[tid] = __cosf(time_b[tid]);
  __syncthreads();
  long id = idS[lm];
  const float* nra = nr + id * (long)Dd;
  const float* nea = ne + id * (long)Dd;

  v8f acc0 = {}, acc1 = {};
  for (int kk = 0; kk < 8; ++kk) {
    int kbase = kk << 5;
    v16bf a;
#pragma unroll
    for (int p = 0; p < 8; ++p) {
      int k = kbase + ((p >> 2) << 4) + ((p & 3) << 1) + (half << 3);
      float f0, f1;
      if (k < 128) { f0 = nra[k] + nea[k]; f1 = nra[k + 1] + nea[k + 1]; }
      else { f0 = tcS[k - 128]; f1 = tcS[k - 127]; }
      a[2 * p] = (__bf16)f0; a[2 * p + 1] = (__bf16)f1;
    }
    const __bf16* bb = Wq_sw + ((long)(kk * 16 + 2 * wv) << 9);
    v16bf b0 = load_bfrag(bb, lane);
    v16bf b1 = load_bfrag(bb + 512, lane);
    acc0 = wmma_bf16(a, b0, acc0);
    acc1 = wmma_bf16(a, b1, acc1);
  }
#pragma unroll
  for (int j = 0; j < 8; ++j) {
    int m = j + (half << 3);
    long row = m0 + m;
    qout[row * 256 + wv * 32 + lm]      = acc0[j];
    qout[row * 256 + wv * 32 + 16 + lm] = acc1[j];
  }
}

// ---------------------------------------------------------------------------
// Kernel 2: fused attention per center (one workgroup = one center):
//   build feat[20x484] in LDS (gathers + time-enc + importance MLP),
//   K = feat@Wk, V = feat@Wv via WMMA (K/V stay in accumulators),
//   scores via LDS f32 atomics, masked softmax, ctx = attn@V -> ctx_out bf16.
// ---------------------------------------------------------------------------
__global__ __launch_bounds__(256) void aggregate(
    const float* __restrict__ qbuf,
    const int* __restrict__ nids, const int* __restrict__ eids,
    const float* __restrict__ nts, const float* __restrict__ ndeg,
    const float* __restrict__ nre, const float* __restrict__ ts, int tsDiv,
    const float* __restrict__ nr, const float* __restrict__ ne,
    const float* __restrict__ h1, int useH1,
    const float* __restrict__ ee,
    const float* __restrict__ time_w, const float* __restrict__ time_b,
    const float* __restrict__ iw1, const float* __restrict__ ib1,
    const float* __restrict__ iw2, const float* __restrict__ ib2,
    const __bf16* __restrict__ Wk_sw, const __bf16* __restrict__ Wv_sw,
    __bf16* __restrict__ ctx_out) {
  __shared__ __bf16 feat[32 * FS];
  __shared__ float qs[256];
  __shared__ float scoreS[32];
  __shared__ float attnS[32];
  __shared__ int nidS[20], eidS[20];
  __shared__ float dtS[20], sS[20];

  int i = blockIdx.x;
  int tid = threadIdx.x;

  // zero feat tile (covers pad rows 20..31 and pad cols 484..511)
  unsigned int* fz = reinterpret_cast<unsigned int*>(feat);
  for (int e = tid; e < 32 * FS / 2; e += 256) fz[e] = 0u;
  qs[tid] = qbuf[(long)i * 256 + tid];
  if (tid < 32) scoreS[tid] = 0.0f;
  if (tid < Kn) {
    nidS[tid] = nids[i * Kn + tid];
    eidS[tid] = eids[i * Kn + tid];
    dtS[tid]  = ts[i / tsDiv] - nts[i * Kn + tid];
    float re  = nre[i * Kn + tid];
    sS[tid]   = (re == 0.0f) ? 0.0f : (ndeg[i * Kn + tid] / re);
  }
  __syncthreads();

  // ---- build feat rows: [neigh_x(128) | edge_x(128) | edge_tx(128) | imp(100)]
  for (int idx = tid; idx < Kn * FEAT; idx += 256) {
    int r = idx / FEAT, c = idx - r * FEAT;
    float v;
    if (c < 128) {
      v = useH1 ? h1[((long)i * Kn + r) * Dd + c]
                : nr[(long)nidS[r] * Dd + c] + ne[(long)nidS[r] * Dd + c];
    } else if (c < 256) {
      v = ee[(long)eidS[r] * Dd + (c - 128)];
    } else if (c < 384) {
      // |dt*w + b| is small (inputs ~U[0,1] / N(0,1)) -> HW cos fast path
      v = __cosf(fmaf(dtS[r], time_w[c - 256], time_b[c - 256]));
    } else {
      int j2 = c - 384;
      float s = sS[r];
      float sum = ib2[j2];
      for (int j = 0; j < 100; ++j) {
        float hh = fmaf(s, iw1[j], ib1[j]);
        hh = fmaxf(hh, 0.0f);
        sum = fmaf(hh, iw2[j * 100 + j2], sum);
      }
      v = sum;
    }
    feat[r * FS + c] = (__bf16)v;
  }
  __syncthreads();

  // ---- K and V projections via WMMA (wave w owns DK cols [32w, 32w+32))
  int wv = tid >> 5, lane = tid & 31, half = lane >> 4, lm = lane & 15;
  v8f kacc[2][2] = {};
  v8f vacc[2][2] = {};
  const __bf16* rowA0 = feat + (long)lm * FS;
  const __bf16* rowA1 = feat + (long)(16 + lm) * FS;
  for (int kk = 0; kk < 16; ++kk) {
    int kbase = kk << 5;
    v16bf a0 = load_afrag_lds(rowA0, kbase, half);
    v16bf a1 = load_afrag_lds(rowA1, kbase, half);
    const __bf16* bk = Wk_sw + ((long)(kk * 16 + 2 * wv) << 9);
    const __bf16* bv = Wv_sw + ((long)(kk * 16 + 2 * wv) << 9);
    __builtin_prefetch(bk + 512 * 16, 0, 1);   // next K-step fragments
    v16bf bk0 = load_bfrag(bk, lane),       bk1 = load_bfrag(bk + 512, lane);
    v16bf bv0 = load_bfrag(bv, lane),       bv1 = load_bfrag(bv + 512, lane);
    kacc[0][0] = wmma_bf16(a0, bk0, kacc[0][0]);
    kacc[0][1] = wmma_bf16(a0, bk1, kacc[0][1]);
    kacc[1][0] = wmma_bf16(a1, bk0, kacc[1][0]);
    kacc[1][1] = wmma_bf16(a1, bk1, kacc[1][1]);
    vacc[0][0] = wmma_bf16(a0, bv0, vacc[0][0]);
    vacc[0][1] = wmma_bf16(a0, bv1, vacc[0][1]);
    vacc[1][0] = wmma_bf16(a1, bv0, vacc[1][0]);
    vacc[1][1] = wmma_bf16(a1, bv1, vacc[1][1]);
  }

  // ---- scores[m] = sum_d q[d]*K[m][d] via LDS atomics over col slices
#pragma unroll
  for (int mt = 0; mt < 2; ++mt)
#pragma unroll
    for (int nt = 0; nt < 2; ++nt)
#pragma unroll
      for (int j = 0; j < 8; ++j) {
        int m = mt * 16 + j + half * 8;
        if (m < Kn) {
          int n = wv * 32 + nt * 16 + lm;
          atomicAdd(&scoreS[m], kacc[mt][nt][j] * qs[n]);
        }
      }
  __syncthreads();

  if (tid == 0) {  // masked softmax over the 20 neighbors (tiny, serial)
    const float scale = 0.0625f;  // 1/sqrt(256)
    float sc[Kn];
    float mx = -3.0e38f;
    for (int m = 0; m < Kn; ++m) {
      float s = scoreS[m] * scale;
      if (nidS[m] == 0) s = -1.0e9f;
      sc[m] = s;
      mx = fmaxf(mx, s);
    }
    float sum = 0.0f;
    for (int m = 0; m < Kn; ++m) { float e = __expf(sc[m] - mx); sc[m] = e; sum += e; }
    float inv = 1.0f / sum;
    for (int m = 0; m < Kn; ++m) attnS[m] = sc[m] * inv;
    for (int m = Kn; m < 32; ++m) attnS[m] = 0.0f;
  }
  __syncthreads();

  // ---- ctx[n] = sum_m attn[m]*V[m][n]; combine lane halves with shfl_xor(16)
#pragma unroll
  for (int nt = 0; nt < 2; ++nt) {
    float c = 0.0f;
#pragma unroll
    for (int mt = 0; mt < 2; ++mt)
#pragma unroll
      for (int j = 0; j < 8; ++j)
        c += attnS[mt * 16 + j + half * 8] * vacc[mt][nt][j];
    c += __shfl_xor(c, 16, 32);
    if (half == 0)
      ctx_out[(long)i * 256 + wv * 32 + nt * 16 + lm] = (__bf16)c;
  }
}

// ---------------------------------------------------------------------------
// Kernel 3: LSTM-cell step fused into the gates GEMM.  h0==c0==0, so only the
// i/g/o quarters of lstm_wih are computed; activation epilogue writes h_new bf16.
// grid = (n/16, 8); 8 waves per block -> 64 col-tiles of H per M tile.
// ---------------------------------------------------------------------------
__global__ __launch_bounds__(256) void lstm_gates(const __bf16* __restrict__ ctx,
                                                  const float* __restrict__ lstm_b,
                                                  const __bf16* __restrict__ Wih_sw,
                                                  __bf16* __restrict__ hnew) {
  int tid = threadIdx.x;
  int wv = tid >> 5, lane = tid & 31, half = lane >> 4, lm = lane & 15;
  int m0 = blockIdx.x * 16;
  int ct = blockIdx.y * 8 + wv;           // 16-col tile within H (0..63)
  const __bf16* arow = ctx + (long)(m0 + lm) * 256;

  v8f ai = {}, ag = {}, ao = {};
  for (int kk = 0; kk < 8; ++kk) {
    int kbase = kk << 5;
    v16bf a;
#pragma unroll
    for (int p = 0; p < 8; ++p) {
      int k = kbase + ((p >> 2) << 4) + ((p & 3) << 1) + (half << 3);
      BF16x2 c;
      c.u = *reinterpret_cast<const unsigned int*>(arow + k);
      a[2 * p] = c.h[0]; a[2 * p + 1] = c.h[1];
    }
    v16bf bi = load_bfrag(Wih_sw + ((long)(kk * 256 + ct) << 9), lane);
    v16bf bg = load_bfrag(Wih_sw + ((long)(kk * 256 + ct + 128) << 9), lane);
    v16bf bo = load_bfrag(Wih_sw + ((long)(kk * 256 + ct + 192) << 9), lane);
    ai = wmma_bf16(a, bi, ai);
    ag = wmma_bf16(a, bg, ag);
    ao = wmma_bf16(a, bo, ao);
  }
  int col = ct * 16 + lm;
  float b_i = lstm_b[col], b_g = lstm_b[col + 2048], b_o = lstm_b[col + 3072];
#pragma unroll
  for (int j = 0; j < 8; ++j) {
    int m = j + half * 8;
    float gi = ai[j] + b_i;
    float gg = ag[j] + b_g;
    float go = ao[j] + b_o;
    float cn = fast_sigmoid(gi) * tanhf(gg);
    float hv = fast_sigmoid(go) * tanhf(cn);
    hnew[(long)(m0 + m) * 1024 + col] = (__bf16)hv;
  }
}

// ---------------------------------------------------------------------------
// Kernel 4: out = [node_feats(ids) , h_new] @ Wout  (concat fused into A load)
// grid = n/16; wave w owns N-tile w (N=128); K = 1152 (36 steps of 32).
// ---------------------------------------------------------------------------
__global__ __launch_bounds__(256) void gemm_out(const int* __restrict__ ids,
                                                const float* __restrict__ nr,
                                                const float* __restrict__ ne,
                                                const __bf16* __restrict__ hnew,
                                                const __bf16* __restrict__ Wout_sw,
                                                float* __restrict__ out) {
  __shared__ int idS[16];
  int tid = threadIdx.x;
  int wv = tid >> 5, lane = tid & 31, half = lane >> 4, lm = lane & 15;
  int m0 = blockIdx.x * 16;
  if (tid < 16) idS[tid] = ids[m0 + tid];
  __syncthreads();
  long id = idS[lm];
  const float* nra = nr + id * (long)Dd;
  const float* nea = ne + id * (long)Dd;
  const __bf16* hrow = hnew + (long)(m0 + lm) * 1024;

  v8f acc = {};
  for (int kk = 0; kk < 36; ++kk) {
    int kbase = kk << 5;
    v16bf a;
#pragma unroll
    for (int p = 0; p < 8; ++p) {
      int k = kbase + ((p >> 2) << 4) + ((p & 3) << 1) + (half << 3);
      if (k < 128) {
        a[2 * p]     = (__bf16)(nra[k] + nea[k]);
        a[2 * p + 1] = (__bf16)(nra[k + 1] + nea[k + 1]);
      } else {
        BF16x2 c;
        c.u = *reinterpret_cast<const unsigned int*>(hrow + (k - 128));
        a[2 * p] = c.h[0]; a[2 * p + 1] = c.h[1];
      }
    }
    v16bf b = load_bfrag(Wout_sw + ((long)(kk * 8 + wv) << 9), lane);
    acc = wmma_bf16(a, b, acc);
  }
#pragma unroll
  for (int j = 0; j < 8; ++j) {
    int m = j + half * 8;
    out[(long)(m0 + m) * 128 + wv * 16 + lm] = acc[j];
  }
}

// ---------------------------------------------------------------------------
// Host launch
// ---------------------------------------------------------------------------
extern "C" void kernel_launch(void* const* d_in, const int* in_sizes, int n_in,
                              void* d_out, int out_size, void* d_ws, size_t ws_size,
                              hipStream_t stream) {
  (void)in_sizes; (void)n_in; (void)out_size; (void)ws_size;

  const float* node_reprs = (const float*)d_in[0];
  const float* node_embed = (const float*)d_in[1];
  const float* edge_embed = (const float*)d_in[2];
  const float* time_w = (const float*)d_in[3];
  const float* time_b = (const float*)d_in[4];
  const float* imp_w1 = (const float*)d_in[5];
  const float* imp_b1 = (const float*)d_in[6];
  const float* imp_w2 = (const float*)d_in[7];
  const float* imp_b2 = (const float*)d_in[8];
  const float* Wq   = (const float*)d_in[9];
  const float* Wk   = (const float*)d_in[10];
  const float* Wv   = (const float*)d_in[11];
  const float* Wih  = (const float*)d_in[12];
  // d_in[13] = lstm_whh: unused, h0 == 0
  const float* lstm_b = (const float*)d_in[14];
  const float* Wout = (const float*)d_in[15];
  const float* ts    = (const float*)d_in[16];
  const float* n2_ts = (const float*)d_in[17];
  const float* n1_ts = (const float*)d_in[18];
  const float* n2_re = (const float*)d_in[19];
  const float* n2_deg = (const float*)d_in[20];
  const float* n1_re  = (const float*)d_in[21];
  const float* n1_deg = (const float*)d_in[22];
  const int* center_nids = (const int*)d_in[23];
  const int* n2_nids = (const int*)d_in[24];
  const int* n2_eids = (const int*)d_in[25];
  const int* n1_nids = (const int*)d_in[26];
  const int* n1_eids = (const int*)d_in[27];

  const int N1 = Bq * Kn;  // 5120 depth-1 centers

  char* p = (char*)d_ws;
  __bf16* sw = (__bf16*)p;           p += TOTAL_SW * 2;
  float* q1  = (float*)p;            p += (long)N1 * 256 * 4;
  float* q2  = (float*)p;            p += (long)Bq * 256 * 4;
  float* h1  = (float*)p;            p += (long)N1 * 128 * 4;
  __bf16* ctx1 = (__bf16*)p;         p += (long)N1 * 256 * 2;
  __bf16* ctx2 = (__bf16*)p;         p += (long)Bq * 256 * 2;
  __bf16* hn1  = (__bf16*)p;         p += (long)N1 * 1024 * 2;
  __bf16* hn2  = (__bf16*)p;         p += (long)Bq * 1024 * 2;

  // 0) weights -> bf16, pre-swizzled B-fragment layout
  prep_weights<<<(int)((TOTAL_SW + 255) / 256), 256, 0, stream>>>(
      Wq, Wk, Wv, Wih, Wout, sw);

  // 1) q projections for both depths
  gemm_q<<<N1 / 16, 256, 0, stream>>>(n2_nids, node_reprs, node_embed, time_b,
                                      sw + OFF_WQ, q1);
  gemm_q<<<Bq / 16, 256, 0, stream>>>(center_nids, node_reprs, node_embed, time_b,
                                      sw + OFF_WQ, q2);

  // 2) depth-1 fused attention
  aggregate<<<N1, 256, 0, stream>>>(q1, n1_nids, n1_eids, n1_ts, n1_deg, n1_re,
                                    ts, Kn, node_reprs, node_embed, nullptr, 0,
                                    edge_embed, time_w, time_b,
                                    imp_w1, imp_b1, imp_w2, imp_b2,
                                    sw + OFF_WK, sw + OFF_WV, ctx1);
  lstm_gates<<<dim3(N1 / 16, 8), 256, 0, stream>>>(ctx1, lstm_b, sw + OFF_WIH, hn1);
  gemm_out<<<N1 / 16, 256, 0, stream>>>(n2_nids, node_reprs, node_embed, hn1,
                                        sw + OFF_WOUT, h1);

  // 3) depth-2 fused attention (neighbor features = h1)
  aggregate<<<Bq, 256, 0, stream>>>(q2, n2_nids, n2_eids, n2_ts, n2_deg, n2_re,
                                    ts, 1, node_reprs, node_embed, h1, 1,
                                    edge_embed, time_w, time_b,
                                    imp_w1, imp_b1, imp_w2, imp_b2,
                                    sw + OFF_WK, sw + OFF_WV, ctx2);
  lstm_gates<<<dim3(Bq / 16, 8), 256, 0, stream>>>(ctx2, lstm_b, sw + OFF_WIH, hn2);
  gemm_out<<<Bq / 16, 256, 0, stream>>>(center_nids, node_reprs, node_embed, hn2,
                                        sw + OFF_WOUT, (float*)d_out);
}
